// VectorQuantizer_55920474194574
// MI455X (gfx1250) — compile-verified, matrix operand order checked by reading, not run on hardware
//
#include <hip/hip_runtime.h>
#include <hip/hip_bf16.h>

typedef __bf16        v16bf __attribute__((ext_vector_type(16)));
typedef float         v8f   __attribute__((ext_vector_type(8)));
typedef float         v4f   __attribute__((ext_vector_type(4)));
typedef unsigned int  u32x4 __attribute__((ext_vector_type(4)));
typedef int           i32x8 __attribute__((ext_vector_type(8)));
typedef int           i32x4 __attribute__((ext_vector_type(4)));

#define N_PTS 32768
#define K_CB  8192
#define D_DIM 512

#define M_TILE   64
#define WAVES    4
#define THREADS  (WAVES * 32)
#define CT       16                       // codebook columns per tile
#define NTILES   (K_CB / CT)              // 512
#define TILE_BF16  16384                  // per-tile bf16 elems (hi 8192 + lo 8192) = 32 KB
#define TILE_BYTES 32768

// ---------------- Kernel A: ||w_k||^2 ----------------
__global__ void __launch_bounds__(128)
wnorm_kernel(const float* __restrict__ cb, float* __restrict__ wnorm) {
    int k = blockIdx.x;
    const float* row = cb + (size_t)k * D_DIM;
    float s = 0.f;
    for (int i = threadIdx.x; i < D_DIM; i += 128) { float v = row[i]; s += v * v; }
    for (int m = 16; m >= 1; m >>= 1) s += __shfl_xor(s, m, 32);
    __shared__ float red[4];
    if ((threadIdx.x & 31) == 0) red[threadIdx.x >> 5] = s;
    __syncthreads();
    if (threadIdx.x == 0) wnorm[k] = red[0] + red[1] + red[2] + red[3];
}

// ------ Kernel A2: one-time codebook swizzle into B-fragment layout (bf16 hi/lo) ------
__global__ void __launch_bounds__(128)
swizzle_cb_kernel(const float* __restrict__ cb, __bf16* __restrict__ cbswz) {
    const int t    = blockIdx.x;
    const int wave = threadIdx.x >> 5;
    const int lane = threadIdx.x & 31;
    const int g    = lane >> 4;
    const int l    = lane & 15;
    const float* wr = cb + (size_t)(t * CT + l) * D_DIM;
    __bf16* tile = cbswz + (size_t)t * TILE_BF16;
    #pragma unroll
    for (int cc = 0; cc < 4; ++cc) {
        const int c = wave * 4 + cc;
        const float* p = wr + c * 32 + 8 * g;
        float f[16];
        *(v4f*)(f + 0)  = *(const v4f*)(p + 0);
        *(v4f*)(f + 4)  = *(const v4f*)(p + 4);
        *(v4f*)(f + 8)  = *(const v4f*)(p + 16);
        *(v4f*)(f + 12) = *(const v4f*)(p + 20);
        v16bf vh, vl;
        #pragma unroll
        for (int e = 0; e < 16; ++e) {
            __bf16 h = (__bf16)f[e];
            vh[e] = h;
            vl[e] = (__bf16)(f[e] - (float)h);
        }
        *(v16bf*)(tile + (size_t)c * 512 + lane * 16)        = vh;
        *(v16bf*)(tile + 8192 + (size_t)c * 512 + lane * 16) = vl;
    }
}

// ---- TDM: DMA one contiguous 32 KB tile (as 4096 x 8B, single row) into LDS ----
__device__ __forceinline__ void tdm_load_tile(const void* gaddr, unsigned lds_off) {
    unsigned long long ga = (unsigned long long)gaddr;
    u32x4 g0;
    g0[0] = 1u;                                    // count=1, user mode
    g0[1] = lds_off;                               // lds_addr (bytes)
    g0[2] = (unsigned)ga;                          // global_addr[31:0]
    g0[3] = (unsigned)((ga >> 32) & 0x01FFFFFFu)   // global_addr[56:32]
          | (2u << 30);                            // type=2 (image)
    i32x8 g1;
    g1[0] = 0x00030000;                            // data_size=3 (8B), wg_mask=0
    g1[1] = (int)(0x1000u << 16);                  // tensor_dim0[15:0]=4096 @ bits 63:48
    g1[2] = 0x00010000;                            // tensor_dim1=1 @ bits 95:80
    g1[3] = (int)(0x1000u << 16);                  // tile_dim0=4096 @ bits 127:112
    g1[4] = 1;                                     // tile_dim1=1
    g1[5] = 4096;                                  // tensor_dim0_stride[31:0]
    g1[6] = 0;
    g1[7] = 0;
    i32x4 z4 = {0, 0, 0, 0};
    i32x8 z8 = {0, 0, 0, 0, 0, 0, 0, 0};
    __builtin_amdgcn_tensor_load_to_lds(g0, g1, z4, z4, z8, 0);
}

// ------- Kernel B: fused x@cbT GEMM (bf16x3 WMMA) + row argmin -------
__global__ void __launch_bounds__(THREADS, 1)
argmin_kernel(const float* __restrict__ x, const __bf16* __restrict__ cbswz,
              const float* __restrict__ wnorm, int* __restrict__ out_idx) {
    extern __shared__ __bf16 lds[];            // 2 * TILE_BF16 = 64 KB, double-buffered
    const int tid  = threadIdx.x;
    const int wave = tid >> 5;
    const int lane = tid & 31;
    const int g    = lane >> 4;
    const int l    = lane & 15;
    const int rowBase = blockIdx.x * M_TILE;
    const int myRow   = rowBase + wave * 16 + l;

    // ---- A fragments: load fp32 row once, split bf16 hi/lo, keep resident ----
    v16bf a_hi[16], a_lo[16];
    {
        const float* xr = x + (size_t)myRow * D_DIM;
        #pragma unroll
        for (int c = 0; c < 16; ++c) {
            const float* p = xr + c * 32 + 8 * g;
            float f[16];
            *(v4f*)(f + 0)  = *(const v4f*)(p + 0);
            *(v4f*)(f + 4)  = *(const v4f*)(p + 4);
            *(v4f*)(f + 8)  = *(const v4f*)(p + 16);
            *(v4f*)(f + 12) = *(const v4f*)(p + 20);
            #pragma unroll
            for (int e = 0; e < 16; ++e) {
                __bf16 h = (__bf16)f[e];
                a_hi[c][e] = h;
                a_lo[c][e] = (__bf16)(f[e] - (float)h);
            }
        }
    }

    float minv[8];
    int   mini[8];
    #pragma unroll
    for (int v = 0; v < 8; ++v) { minv[v] = 3.402823466e+38f; mini[v] = 0; }

    const char* gbase = (const char*)cbswz;

    // Prologue: DMA tile 0 into buffer 0 (wave 0 issues; TDM ignores EXEC)
    if (wave == 0) tdm_load_tile(gbase, 0);

    for (int t = 0; t < NTILES; ++t) {
        const int colBase = t * CT;
        if (wave == 0) __builtin_amdgcn_s_wait_tensorcnt(0);
        __syncthreads();                       // tile t ready; prev buffer free
        if (wave == 0 && t + 1 < NTILES)
            tdm_load_tile(gbase + (size_t)(t + 1) * TILE_BYTES,
                          (unsigned)(((t + 1) & 1) * TILE_BYTES));

        const __bf16* buf = lds + (t & 1) * TILE_BF16;
        float wn = wnorm[colBase + l];
        v8f acc0 = {0.f, 0.f, 0.f, 0.f, 0.f, 0.f, 0.f, 0.f};
        v8f acc1 = {0.f, 0.f, 0.f, 0.f, 0.f, 0.f, 0.f, 0.f};

        // B-fragment software pipeline: loads for chunk c+1 issued before chunk c's WMMAs
        v16bf bh = *(const v16bf*)(buf + lane * 16);
        v16bf bl = *(const v16bf*)(buf + 8192 + lane * 16);
        #pragma unroll
        for (int c = 0; c < 16; ++c) {
            const int cn = (c + 1) & 15;       // clamp: last prefetch is unused
            v16bf bh_n = *(const v16bf*)(buf + cn * 512 + lane * 16);
            v16bf bl_n = *(const v16bf*)(buf + 8192 + cn * 512 + lane * 16);
            acc0 = __builtin_amdgcn_wmma_f32_16x16x32_bf16(false, a_hi[c], false, bh,
                                                           (short)0, acc0, false, false);
            acc1 = __builtin_amdgcn_wmma_f32_16x16x32_bf16(false, a_lo[c], false, bh,
                                                           (short)0, acc1, false, false);
            if (c & 1)
                acc1 = __builtin_amdgcn_wmma_f32_16x16x32_bf16(false, a_hi[c], false, bl,
                                                               (short)0, acc1, false, false);
            else
                acc0 = __builtin_amdgcn_wmma_f32_16x16x32_bf16(false, a_hi[c], false, bl,
                                                               (short)0, acc0, false, false);
            bh = bh_n;
            bl = bl_n;
        }

        // C/D layout: element v = (row v+8g, col l)
        #pragma unroll
        for (int v = 0; v < 8; ++v) {
            float dist = wn - 2.0f * (acc0[v] + acc1[v]);
            if (dist < minv[v]) { minv[v] = dist; mini[v] = colBase + l; }
        }
    }

    // argmin across the 16 lanes of each group (columns); lowest index wins ties
    #pragma unroll
    for (int v = 0; v < 8; ++v) {
        float mv = minv[v];
        int   mi = mini[v];
        #pragma unroll
        for (int m = 8; m >= 1; m >>= 1) {
            float ov = __shfl_xor(mv, m, 16);
            int   oi = __shfl_xor(mi, m, 16);
            if (ov < mv || (ov == mv && oi < mi)) { mv = ov; mi = oi; }
        }
        if (l == 0) out_idx[rowBase + wave * 16 + g * 8 + v] = mi;
    }
}

// ---------------- Kernel C: zero-fill encodings ----------------
__global__ void fill_zero_kernel(v4f* __restrict__ p, size_t n4) {
    size_t i = (size_t)blockIdx.x * blockDim.x + threadIdx.x;
    size_t stride = (size_t)gridDim.x * blockDim.x;
    v4f z = {0.f, 0.f, 0.f, 0.f};
    for (; i < n4; i += stride) p[i] = z;
}

// ------- Kernel D: gather quantized rows, scatter one-hot, indices -------
__global__ void __launch_bounds__(128)
gather_kernel(const float* __restrict__ cb, const int* __restrict__ idx,
              float* __restrict__ quant, float* __restrict__ enc,
              float* __restrict__ idx_out) {
    int n = blockIdx.x;
    int k = idx[n];
    const float* src = cb + (size_t)k * D_DIM;
    float* dst = quant + (size_t)n * D_DIM;
    *(v4f*)(dst + threadIdx.x * 4) = *(const v4f*)(src + threadIdx.x * 4);
    if (threadIdx.x == 0) {
        enc[(size_t)n * K_CB + k] = 1.0f;
        idx_out[n] = (float)k;
    }
}

extern "C" void kernel_launch(void* const* d_in, const int* in_sizes, int n_in,
                              void* d_out, int out_size, void* d_ws, size_t ws_size,
                              hipStream_t stream) {
    const float* x  = (const float*)d_in[0];   // [N, D] fp32
    const float* cb = (const float*)d_in[1];   // [K, D] fp32

    float* quant   = (float*)d_out;                              // [N, D]
    float* enc     = quant + (size_t)N_PTS * D_DIM;              // [N, K]
    float* idx_out = enc + (size_t)N_PTS * K_CB;                 // [N]

    // ws layout: wnorm (32 KB) | idx (128 KB) | swizzled codebook (16 MB)
    float*  wnorm = (float*)d_ws;
    int*    idx   = (int*)((char*)d_ws + (size_t)K_CB * sizeof(float));
    __bf16* cbswz = (__bf16*)((char*)d_ws + 160 * 1024);

    wnorm_kernel<<<K_CB, 128, 0, stream>>>(cb, wnorm);
    swizzle_cb_kernel<<<NTILES, 128, 0, stream>>>(cb, cbswz);

    size_t ldsBytes = 2 * (size_t)TILE_BF16 * sizeof(__bf16);    // 64 KB double buffer
    argmin_kernel<<<N_PTS / M_TILE, THREADS, ldsBytes, stream>>>(x, cbswz, wnorm, idx);

    fill_zero_kernel<<<4096, 256, 0, stream>>>((v4f*)enc, (size_t)N_PTS * K_CB / 4);

    gather_kernel<<<N_PTS, 128, 0, stream>>>(cb, idx, quant, enc, idx_out);
}
